// AG_RoPE_EncoderLayer_32658931319249
// MI455X (gfx1250) — compile-verified, matrix operand order checked by reading, not run on hardware
//
#include <hip/hip_runtime.h>

// ---------------------------------------------------------------------------
// CDNA5 / gfx1250 implementation of the LoFTR-style encoder layer.
// All heavy GEMMs use v_wmma_f32_16x16x32_bf16 (wave32).
// Round 2: LDS-staged A tiles (async global->LDS), 2-4 accumulators per wave
// for A-fragment reuse, near-cache prefetch.
// ---------------------------------------------------------------------------

typedef __attribute__((ext_vector_type(16))) __bf16 v16bf;
typedef __attribute__((ext_vector_type(8)))  float  v8f;

#define Z8 {0.f, 0.f, 0.f, 0.f, 0.f, 0.f, 0.f, 0.f}

union FragBF {
    v16bf v;
    uint4 u[2];
    unsigned short e[16];
};

__device__ __forceinline__ unsigned short f2bf(float f) {
    union { float f; unsigned u; } c; c.f = f;
    unsigned u = c.u;
    unsigned lsb = (u >> 16) & 1u;
    u += 0x7fffu + lsb;                 // round-to-nearest-even
    return (unsigned short)(u >> 16);
}
__device__ __forceinline__ float bf2f(unsigned short h) {
    union { unsigned u; float f; } c; c.u = ((unsigned)h) << 16;
    return c.f;
}

__device__ __forceinline__ v8f wmma_bf16(v16bf a, v16bf b, v8f c) {
    // D = A(16x32 bf16) * B(32x16 bf16) + C(16x16 f32)
    return __builtin_amdgcn_wmma_f32_16x16x32_bf16(
        /*neg_a=*/false, a, /*neg_b=*/false, b,
        /*c_mod=*/(short)0, c, /*reuse_a=*/false, /*reuse_b=*/false);
}

// Row-major fragment load (A rows, or B via pre-transposed W^T [N][K]):
// lane group g holds K in {k0+8g .. k0+8g+7} and {k0+16+8g .. k0+16+8g+7}.
__device__ __forceinline__ v16bf ld_frag(const unsigned short* __restrict__ base,
                                         int r0, int k0, int ld) {
    const int lane = threadIdx.x & 31;
    const int m = lane & 15;
    const int g = lane >> 4;
    const unsigned short* p = base + (size_t)(r0 + m) * ld + (k0 + (g << 3));
    FragBF f;
    f.u[0] = *(const uint4*)(p);
    f.u[1] = *(const uint4*)(p + 16);
    return f.v;
}

// Strided (gather) fragment: element (m, k) at base[idx0 + m + k*kstride].
__device__ __forceinline__ v16bf ld_frag_strided(const unsigned short* __restrict__ base,
                                                 int idx0, int k0, int kstride) {
    const int lane = threadIdx.x & 31;
    const int m = lane & 15;
    const int g = lane >> 4;
    const unsigned short* p = base + idx0 + m;
    FragBF f;
#pragma unroll
    for (int i = 0; i < 8; ++i)
        f.e[i] = p[(size_t)(k0 + (g << 3) + i) * kstride];
#pragma unroll
    for (int i = 0; i < 8; ++i)
        f.e[8 + i] = p[(size_t)(k0 + 16 + (g << 3) + i) * kstride];
    return f.v;
}

// ---------------------------------------------------------------------------
// Async global->LDS staging (ASYNCcnt path, CDNA5 ISA).
// VDST VGPR = LDS byte offset, VADDR = 64-bit global address.
// ---------------------------------------------------------------------------
#define USE_ASYNC_LDS 1

typedef __attribute__((address_space(3))) void lds_void_t;

__device__ __forceinline__ void async_copy16(const void* g, void* l) {
#if USE_ASYNC_LDS
    unsigned loff = (unsigned)(unsigned long long)(lds_void_t*)l; // LDS byte offset
    unsigned long long ga = (unsigned long long)g;
    asm volatile("global_load_async_to_lds_b128 %0, %1, off"
                 :: "v"(loff), "v"(ga) : "memory");
#else
    *(uint4*)l = *(const uint4*)g;
#endif
}
__device__ __forceinline__ void async_join() {
#if USE_ASYNC_LDS
    asm volatile("s_wait_asynccnt 0x0" ::: "memory");
#endif
}

// ---------------------------------------------------------------------------
// K0: weight f32[K][N] -> bf16 transposed Wt[N][K]
// ---------------------------------------------------------------------------
__global__ void wt_bf16_kernel(const float* __restrict__ W,
                               unsigned short* __restrict__ Wt, int K, int N) {
    int i = blockIdx.x * 256 + threadIdx.x;
    if (i < K * N) {
        int k = i / N, n = i - k * N;
        Wt[(size_t)n * K + k] = f2bf(W[i]);
    }
}

// ---------------------------------------------------------------------------
// K1: maxpool 4x4 (NCHW) -> NHWC row -> layernorm -> bf16 [4096][256]
// ---------------------------------------------------------------------------
__global__ __launch_bounds__(256) void pool_ln_kernel(
    const float* __restrict__ src, unsigned short* __restrict__ dst,
    const float* __restrict__ gam, const float* __restrict__ bet) {
    const int c = threadIdx.x;
    const int r = blockIdx.x;                 // 0..4095
    const int b  = r >> 10;
    const int px = r & 1023;
    const int ph = px >> 5, pw = px & 31;
    const float* p = src + (((size_t)(b * 256 + c) * 128 + ph * 4) * 128 + pw * 4);
    float m = -INFINITY;
#pragma unroll
    for (int i = 0; i < 4; ++i)
#pragma unroll
        for (int j = 0; j < 4; ++j)
            m = fmaxf(m, p[i * 128 + j]);

    __shared__ float red[256];
    __shared__ float smean, svar;
    red[c] = m; __syncthreads();
    for (int s = 128; s > 0; s >>= 1) {
        if (c < s) red[c] += red[c + s];
        __syncthreads();
    }
    if (c == 0) smean = red[0] * (1.0f / 256.0f);
    __syncthreads();
    float d = m - smean;
    red[c] = d * d; __syncthreads();
    for (int s = 128; s > 0; s >>= 1) {
        if (c < s) red[c] += red[c + s];
        __syncthreads();
    }
    if (c == 0) svar = red[0] * (1.0f / 256.0f);
    __syncthreads();
    dst[(size_t)r * 256 + c] = f2bf(d * rsqrtf(svar + 1e-5f) * gam[c] + bet[c]);
}

// ---------------------------------------------------------------------------
// K2: WMMA GEMM, fixed N=K=256: out[M,256] = A[M,256] @ Bt[256,256]^T
// One block = 16 rows. A tile staged once into LDS (async), 8 waves each own
// 2 column tiles (2 accumulators, shared A fragment per k-step).
// mode: 0 bf16 raw, 1 bf16 elu(x)+1, 2 bf16 x*scale, 3 f32 raw
// ---------------------------------------------------------------------------
#define GK 256
#define A_LD 264
__global__ __launch_bounds__(256) void gemm256_kernel(
    const unsigned short* __restrict__ A, const unsigned short* __restrict__ Bt,
    void* __restrict__ out, int mode, float scale) {
    __shared__ unsigned short alds[16 * A_LD];

    const int t = threadIdx.x;
    const int row0 = blockIdx.x * 16;

    // stage 16x256 A tile (contiguous 8KB) -> LDS, padded rows
    const unsigned short* gsrc = A + (size_t)row0 * GK;
#pragma unroll
    for (int j = 0; j < 2; ++j) {
        int i = t + 256 * j;              // uint4 index 0..511
        int r = i >> 5;                   // (i*8)/256
        int c = (i << 3) & 255;
        async_copy16(gsrc + i * 8, &alds[r * A_LD + c]);
    }
    async_join();
    __syncthreads();

    const int w = t >> 5;
    const int col0 = w * 32;
    v8f acc0 = Z8, acc1 = Z8;
    for (int k0 = 0; k0 < GK; k0 += 32) {
        if (k0 + 32 < GK)
            __builtin_prefetch(Bt + (size_t)col0 * GK + k0 + 32, 0, 3);
        v16bf a  = ld_frag(alds, 0, k0, A_LD);
        v16bf b0 = ld_frag(Bt, col0,      k0, GK);
        v16bf b1 = ld_frag(Bt, col0 + 16, k0, GK);
        acc0 = wmma_bf16(a, b0, acc0);
        acc1 = wmma_bf16(a, b1, acc1);
    }

    const int lane = t & 31;
    const int nn = lane & 15, g = lane >> 4;
    v8f accs[2] = {acc0, acc1};
#pragma unroll
    for (int ct = 0; ct < 2; ++ct) {
#pragma unroll
        for (int r = 0; r < 8; ++r) {
            float v = accs[ct][r];
            if (mode == 1) v = (v > 0.f) ? (v + 1.f) : __expf(v);   // elu(x)+1
            else if (mode == 2) v *= scale;
            size_t idx = (size_t)(row0 + (g << 3) + r) * 256 + col0 + ct * 16 + nn;
            if (mode == 3) ((float*)out)[idx] = v;
            else ((unsigned short*)out)[idx] = f2bf(v);
        }
    }
}

// ---------------------------------------------------------------------------
// K3: Ksum[n,h,d] = sum_s Kp[n,s,h,d]
// ---------------------------------------------------------------------------
__global__ __launch_bounds__(256) void ksum_kernel(
    const unsigned short* __restrict__ Kp, float* __restrict__ Ks) {
    const int nh = blockIdx.x;
    const int n = nh >> 3, h = nh & 7;
    const int d = threadIdx.x & 31, ch = threadIdx.x >> 5;
    float s = 0.f;
    for (int i = 0; i < 128; ++i) {
        int sidx = ch * 128 + i;
        s += bf2f(Kp[(size_t)(n * 1024 + sidx) * 256 + h * 32 + d]);
    }
    __shared__ float red[8][32];
    red[ch][d] = s; __syncthreads();
    if (threadIdx.x < 32) {
        float t = 0.f;
#pragma unroll
        for (int j = 0; j < 8; ++j) t += red[j][threadIdx.x];
        Ks[nh * 32 + threadIdx.x] = t;
    }
}

// ---------------------------------------------------------------------------
// K4: KV^T[n,h,v,d] = sum_s Kp[n,s,h,d] * Vn[n,s,h,v]  (K = S = 1024)
// ---------------------------------------------------------------------------
__global__ __launch_bounds__(128) void kv_kernel(
    const unsigned short* __restrict__ Kp, const unsigned short* __restrict__ Vn,
    unsigned short* __restrict__ kvT) {
    const int nh = blockIdx.x;
    const int n = nh >> 3, h = nh & 7;
    const int w = threadIdx.x >> 5;
    const int r0 = (w >> 1) * 16;             // d tile
    const int c0 = (w & 1) * 16;              // v tile
    const unsigned short* Kb = Kp + (size_t)n * 1024 * 256 + h * 32;
    const unsigned short* Vb = Vn + (size_t)n * 1024 * 256 + h * 32;

    v8f acc = Z8;
    for (int k0 = 0; k0 < 1024; k0 += 32) {
        v16bf a = ld_frag_strided(Kb, r0, k0, 256);   // A[d][s]
        v16bf b = ld_frag_strided(Vb, c0, k0, 256);   // B[s][v]
        acc = wmma_bf16(a, b, acc);
    }
    const int lane = threadIdx.x & 31;
    const int nn = lane & 15, g = lane >> 4;
#pragma unroll
    for (int r = 0; r < 8; ++r) {
        int d = r0 + (g << 3) + r;
        int v = c0 + nn;
        kvT[(size_t)nh * 1024 + v * 32 + d] = f2bf(acc[r]);   // store [v][d]
    }
}

// ---------------------------------------------------------------------------
// K5: Z[l,h] = 1 / (Qp[l,h,:] . Ksum[n,h,:] + eps)
// ---------------------------------------------------------------------------
__global__ __launch_bounds__(256) void z_kernel(
    const unsigned short* __restrict__ Qp, const float* __restrict__ Ks,
    float* __restrict__ Z) {
    int gid = blockIdx.x * 256 + threadIdx.x;  // 32768 = 4096 * 8
    int l = gid >> 3, h = gid & 7;
    int n = l >> 10;
    float dot = 0.f;
#pragma unroll
    for (int d = 0; d < 32; ++d)
        dot += bf2f(Qp[(size_t)l * 256 + h * 32 + d]) * Ks[(n * 8 + h) * 32 + d];
    Z[gid] = 1.0f / (dot + 1e-6f);
}

// ---------------------------------------------------------------------------
// K6: msg[l, h*32+v] = (Qp[l,h,:] @ KV[:, v]) * Z[l,h] * L
// ---------------------------------------------------------------------------
__global__ __launch_bounds__(256) void attn_kernel(
    const unsigned short* __restrict__ Qp, const unsigned short* __restrict__ kvT,
    const float* __restrict__ Z, unsigned short* __restrict__ msg) {
    const int h = blockIdx.y;
    const int l0 = blockIdx.x * 64;
    const int n = l0 >> 10;
    const int w = threadIdx.x >> 5;
    const int row0 = l0 + (w >> 1) * 16;
    const int c0 = (w & 1) * 16;

    v16bf a = ld_frag(Qp, row0, h * 32, 256);                       // A[l][d]
    v16bf b = ld_frag(kvT + (size_t)(n * 8 + h) * 1024, c0, 0, 32); // Bt[v][d]
    v8f acc = Z8;
    acc = wmma_bf16(a, b, acc);

    const int lane = threadIdx.x & 31;
    const int nn = lane & 15, g = lane >> 4;
#pragma unroll
    for (int r = 0; r < 8; ++r) {
        int l = row0 + (g << 3) + r;
        float v = acc[r] * Z[l * 8 + h] * 1024.0f;
        msg[(size_t)l * 256 + h * 32 + c0 + nn] = f2bf(v);
    }
}

// ---------------------------------------------------------------------------
// K7: fused  cat([x, bilerp(msgm)]) -> relu(cat@w1) -> @w2 -> LN -> +x
// 16 pixels/block, 8 waves. Stage1: 4 accumulators/wave sharing one A frag
// per k-step (N=512,K=512). Stage2: 2 accumulators/wave (N=256,K=512).
// ---------------------------------------------------------------------------
#define CAT_LD 520
__global__ __launch_bounds__(256) void mlp_fused_kernel(
    const float* __restrict__ x, const float* __restrict__ msgm,
    const unsigned short* __restrict__ w1t, const unsigned short* __restrict__ w2t,
    const float* __restrict__ g2, const float* __restrict__ b2,
    float* __restrict__ out) {
    __shared__ unsigned short catb[16][CAT_LD];   // 16 x 512 bf16 (+pad)
    __shared__ unsigned short hidb[16][CAT_LD];
    __shared__ float          outb[16][260];

    const int t = threadIdx.x;

    // ---- build cat tile: [x channels | bilinear-upsampled message channels]
    for (int r = 0; r < 16; ++r) {
        int rr  = blockIdx.x * 16 + r;
        int b   = rr >> 14;
        int pix = rr & 16383;
        int hh  = pix >> 7, ww = pix & 127;

        catb[r][t] = f2bf(x[((size_t)(b * 256 + t) << 14) + pix]);

        // half-pixel bilinear, scale 4, edge clamp
        float sy = (hh + 0.5f) * 0.25f - 0.5f;
        float sx = (ww + 0.5f) * 0.25f - 0.5f;
        int y0 = (int)floorf(sy), x0 = (int)floorf(sx);
        float fy = sy - (float)y0, fx = sx - (float)x0;
        int y0c = min(max(y0, 0), 31), y1c = min(max(y0 + 1, 0), 31);
        int x0c = min(max(x0, 0), 31), x1c = min(max(x0 + 1, 0), 31);
        const float* mp = msgm + (size_t)b * 32 * 32 * 256;
        float v00 = mp[((size_t)y0c * 32 + x0c) * 256 + t];
        float v01 = mp[((size_t)y0c * 32 + x1c) * 256 + t];
        float v10 = mp[((size_t)y1c * 32 + x0c) * 256 + t];
        float v11 = mp[((size_t)y1c * 32 + x1c) * 256 + t];
        float bi = (1.f - fy) * ((1.f - fx) * v00 + fx * v01)
                 +        fy  * ((1.f - fx) * v10 + fx * v11);
        catb[r][256 + t] = f2bf(bi);
    }
    __syncthreads();

    const int w = t >> 5;
    const int lane = t & 31;
    const int nn = lane & 15, g = lane >> 4;

    // ---- stage 1: hidden = relu(cat @ w1)  (N=512), 64 cols per wave
    {
        const int cbase = w * 64;
        v8f acc[4] = {Z8, Z8, Z8, Z8};
        for (int k0 = 0; k0 < 512; k0 += 32) {
            if (k0 + 32 < 512)
                __builtin_prefetch(w1t + (size_t)cbase * 512 + k0 + 32, 0, 3);
            v16bf a = ld_frag(&catb[0][0], 0, k0, CAT_LD);
#pragma unroll
            for (int ct = 0; ct < 4; ++ct) {
                v16bf b = ld_frag(w1t, cbase + ct * 16, k0, 512);
                acc[ct] = wmma_bf16(a, b, acc[ct]);
            }
        }
#pragma unroll
        for (int ct = 0; ct < 4; ++ct)
#pragma unroll
            for (int r = 0; r < 8; ++r) {
                float v = acc[ct][r];
                hidb[(g << 3) + r][cbase + ct * 16 + nn] = f2bf(v > 0.f ? v : 0.f);
            }
    }
    __syncthreads();

    // ---- stage 2: m = hidden @ w2  (N=256), 32 cols per wave
    {
        const int cbase = w * 32;
        v8f acc[2] = {Z8, Z8};
        for (int k0 = 0; k0 < 512; k0 += 32) {
            if (k0 + 32 < 512)
                __builtin_prefetch(w2t + (size_t)cbase * 512 + k0 + 32, 0, 3);
            v16bf a = ld_frag(&hidb[0][0], 0, k0, CAT_LD);
#pragma unroll
            for (int ct = 0; ct < 2; ++ct) {
                v16bf b = ld_frag(w2t, cbase + ct * 16, k0, 512);
                acc[ct] = wmma_bf16(a, b, acc[ct]);
            }
        }
#pragma unroll
        for (int ct = 0; ct < 2; ++ct)
#pragma unroll
            for (int r = 0; r < 8; ++r)
                outb[(g << 3) + r][cbase + ct * 16 + nn] = acc[ct][r];
    }
    __syncthreads();

    // ---- layernorm over 256 channels per row + residual, 2 rows per wave
    for (int rl = 0; rl < 2; ++rl) {
        const int r = w * 2 + rl;
        float vals[8], s = 0.f, s2 = 0.f;
#pragma unroll
        for (int j = 0; j < 8; ++j) {
            float v = outb[r][lane + 32 * j];
            vals[j] = v; s += v; s2 += v * v;
        }
#pragma unroll
        for (int off = 16; off > 0; off >>= 1) {
            s  += __shfl_xor(s,  off, 32);
            s2 += __shfl_xor(s2, off, 32);
        }
        float mean = s * (1.0f / 256.0f);
        float var  = s2 * (1.0f / 256.0f) - mean * mean;
        float rstd = rsqrtf(var + 1e-5f);

        int rr  = blockIdx.x * 16 + r;
        int b   = rr >> 14;
        int pix = rr & 16383;
#pragma unroll
        for (int j = 0; j < 8; ++j) {
            int c = lane + 32 * j;
            float nv = (vals[j] - mean) * rstd * g2[c] + b2[c];
            size_t gi = ((size_t)(b * 256 + c) << 14) + pix;
            out[gi] = x[gi] + nv;
        }
    }
}

// ---------------------------------------------------------------------------
extern "C" void kernel_launch(void* const* d_in, const int* in_sizes, int n_in,
                              void* d_out, int out_size, void* d_ws, size_t ws_size,
                              hipStream_t stream) {
    const float* x       = (const float*)d_in[0];
    const float* source  = (const float*)d_in[1];
    const float* n1g     = (const float*)d_in[2];
    const float* n1b     = (const float*)d_in[3];
    const float* w_q     = (const float*)d_in[4];
    const float* w_k     = (const float*)d_in[5];
    const float* w_v     = (const float*)d_in[6];
    const float* w_merge = (const float*)d_in[7];
    const float* mlp_w1  = (const float*)d_in[8];
    const float* mlp_w2  = (const float*)d_in[9];
    const float* n2g     = (const float*)d_in[10];
    const float* n2b     = (const float*)d_in[11];
    float* out = (float*)d_out;

    char* ws = (char*)d_ws;
    size_t off = 0;
    auto alloc = [&](size_t bytes) { char* p = ws + off; off += (bytes + 255) & ~(size_t)255; return p; };
    unsigned short* wq_t  = (unsigned short*)alloc(256 * 256 * 2);
    unsigned short* wk_t  = (unsigned short*)alloc(256 * 256 * 2);
    unsigned short* wv_t  = (unsigned short*)alloc(256 * 256 * 2);
    unsigned short* wm_t  = (unsigned short*)alloc(256 * 256 * 2);
    unsigned short* w1_t  = (unsigned short*)alloc(512 * 512 * 2);
    unsigned short* w2_t  = (unsigned short*)alloc(256 * 512 * 2);
    unsigned short* q_in  = (unsigned short*)alloc(4096 * 256 * 2);
    unsigned short* s_in  = (unsigned short*)alloc(4096 * 256 * 2);
    unsigned short* Qp    = (unsigned short*)alloc(4096 * 256 * 2);
    unsigned short* Kp    = (unsigned short*)alloc(4096 * 256 * 2);
    unsigned short* Vn    = (unsigned short*)alloc(4096 * 256 * 2);
    unsigned short* msg   = (unsigned short*)alloc(4096 * 256 * 2);
    float*          msgm  = (float*)         alloc(4096 * 256 * 4);
    float*          Ksum  = (float*)         alloc(32 * 32 * 4);
    float*          Z     = (float*)         alloc(32768 * 4);
    unsigned short* kvT   = (unsigned short*)alloc(32 * 32 * 32 * 2);

    // weights -> bf16, transposed [N][K]
    wt_bf16_kernel<<<256,  256, 0, stream>>>(w_q,     wq_t, 256, 256);
    wt_bf16_kernel<<<256,  256, 0, stream>>>(w_k,     wk_t, 256, 256);
    wt_bf16_kernel<<<256,  256, 0, stream>>>(w_v,     wv_t, 256, 256);
    wt_bf16_kernel<<<256,  256, 0, stream>>>(w_merge, wm_t, 256, 256);
    wt_bf16_kernel<<<1024, 256, 0, stream>>>(mlp_w1,  w1_t, 512, 512);
    wt_bf16_kernel<<<512,  256, 0, stream>>>(mlp_w2,  w2_t, 512, 256);

    // maxpool + layernorm
    pool_ln_kernel<<<4096, 256, 0, stream>>>(x,      q_in, n1g, n1b);
    pool_ln_kernel<<<4096, 256, 0, stream>>>(source, s_in, n1g, n1b);

    // projections: Qp = elu(q@wq)+1, Kp = elu(s@wk)+1, Vn = (s@wv)/1024
    gemm256_kernel<<<256, 256, 0, stream>>>(q_in, wq_t, Qp, 1, 0.f);
    gemm256_kernel<<<256, 256, 0, stream>>>(s_in, wk_t, Kp, 1, 0.f);
    gemm256_kernel<<<256, 256, 0, stream>>>(s_in, wv_t, Vn, 2, 1.0f / 1024.0f);

    // linear attention internals
    ksum_kernel<<<32, 256, 0, stream>>>(Kp, Ksum);
    kv_kernel<<<32, 128, 0, stream>>>(Kp, Vn, kvT);
    z_kernel<<<128, 256, 0, stream>>>(Qp, Ksum, Z);
    attn_kernel<<<dim3(64, 8), 256, 0, stream>>>(Qp, kvT, Z, msg);

    // merge projection -> f32 NHWC [4,32,32,256]
    gemm256_kernel<<<256, 256, 0, stream>>>(msg, wm_t, msgm, 3, 0.f);

    // fused upsample + MLP + LN + residual
    mlp_fused_kernel<<<4096, 256, 0, stream>>>(x, msgm, w1_t, w2_t, n2g, n2b, out);
}
